// LSTM_9320079033010
// MI455X (gfx1250) — matrix operand order repaired and checked by reading, not compiled, not run on hardware
//
#include <hip/hip_runtime.h>

#define T_STEPS 1024
#define BATCH   32
#define HID     512
#define GATES   2048   // 4*HID
#define K_DIM   512
#define NBLK    8      // persistent blocks for the recurrent kernel

typedef __attribute__((ext_vector_type(16))) __bf16       v16bf;
typedef __attribute__((ext_vector_type(8)))  float        v8f;
typedef __attribute__((ext_vector_type(4)))  unsigned int u32x4;

union Frag {
    v16bf v;
    unsigned int u[8];
};

__device__ __forceinline__ unsigned short f2bf(float f) {
    unsigned int u = __float_as_uint(f);
    unsigned int r = u + 0x7FFFu + ((u >> 16) & 1u);   // round-to-nearest-even
    return (unsigned short)(r >> 16);
}

__device__ __forceinline__ v8f wmma_bf16(v16bf a, v16bf b, v8f c) {
    // D = A(16x32 bf16) * B(32x16 bf16) + C(16x16 f32)
    return __builtin_amdgcn_wmma_f32_16x16x32_bf16(
        /*neg_a=*/false, a, /*neg_b=*/false, b,
        /*c_mod=*/(short)0, c, /*reuse_a=*/false, /*reuse_b=*/false);
}

// A-matrix 16x32 bf16 fragment. Lane L holds row m = L&15.
// element e -> k = e + 16*(L>=16); loaded as dword pairs
// (32 contiguous bytes -> merged to b128 loads).
__device__ __forceinline__ v16bf load_a_frag(const unsigned short* __restrict__ row,
                                             int kbase, int g) {
    Frag f;
#pragma unroll
    for (int v = 0; v < 8; ++v) {
        int k0 = kbase + 2 * v + 16 * g;
        f.u[v] = *(const unsigned int*)(row + k0);
    }
    return f.v;
}

// B-matrix 32x16 bf16 fragment, B[k][n] = W[n][k] with W row-major:
// lane L holds col n = L&15, k = e + 16*(L>=16) -> 32 contiguous bytes.
__device__ __forceinline__ v16bf load_b_frag(const unsigned short* __restrict__ wrow,
                                             int kbase, int g) {
    Frag f;
    const u32x4* p = (const u32x4*)(wrow + kbase + 16 * g);
    *(u32x4*)(&f.u[0]) = p[0];
    *(u32x4*)(&f.u[4]) = p[1];
    return f.v;
}

// ---------------------------------------------------------------------------
// Big input GEMM: C[r][j] = sum_k A[r][k]*W[j][k] + bias[j]
// A: [32768][512] bf16 (rows r = t*32+b), W: [2048][512] bf16, C fp32.
// Block 256 thr = 8 waves; block tile 128(M) x 64(N); wave tile 64(M) x 16(N).
// ---------------------------------------------------------------------------
__global__ __launch_bounds__(256) void gemm_gates_kernel(
    const unsigned short* __restrict__ A,
    const unsigned short* __restrict__ W,
    const float* __restrict__ bias,
    float* __restrict__ C) {
    const int tid  = threadIdx.x;
    const int wave = tid >> 5;
    const int lane = tid & 31;
    const int g    = lane >> 4;
    const int ln   = lane & 15;
    const int wy   = wave >> 2;                       // 0..1
    const int wx   = wave & 3;                        // 0..3
    const int m_base = blockIdx.y * 128 + wy * 64;
    const int n0     = blockIdx.x * 64  + wx * 16;

    v8f acc[4];
#pragma unroll
    for (int mt = 0; mt < 4; ++mt)
#pragma unroll
        for (int r = 0; r < 8; ++r) acc[mt][r] = 0.0f;

    const unsigned short* wrow = W + (size_t)(n0 + ln) * K_DIM;
#pragma unroll 4
    for (int kk = 0; kk < K_DIM / 32; ++kk) {
        const int kbase = kk * 32;
        v16bf bf = load_b_frag(wrow, kbase, g);
#pragma unroll
        for (int mt = 0; mt < 4; ++mt) {
            const unsigned short* arow = A + (size_t)(m_base + mt * 16 + ln) * K_DIM;
            v16bf af = load_a_frag(arow, kbase, g);
            acc[mt] = wmma_bf16(af, bf, acc[mt]);
        }
    }

    const float bn = bias[n0 + ln];
#pragma unroll
    for (int mt = 0; mt < 4; ++mt)
#pragma unroll
        for (int r = 0; r < 8; ++r) {
            int row = m_base + mt * 16 + 8 * g + r;
            C[(size_t)row * GATES + (n0 + ln)] = acc[mt][r] + bn;
        }
}

// ---------------------------------------------------------------------------
// Device-scope grid barrier for the persistent recurrent kernel.
// Counter increases monotonically; target = NBLK*(t+1).
// ---------------------------------------------------------------------------
__device__ __forceinline__ void grid_sync(unsigned int* __restrict__ counter,
                                          unsigned int target) {
    __threadfence();      // release this lane's global stores (device scope)
    __syncthreads();
    if (threadIdx.x == 0) {
        __hip_atomic_fetch_add(counter, 1u, __ATOMIC_RELEASE, __HIP_MEMORY_SCOPE_AGENT);
        unsigned int v;
        do {
            __builtin_amdgcn_s_sleep(2);
            v = __hip_atomic_load(counter, __ATOMIC_ACQUIRE, __HIP_MEMORY_SCOPE_AGENT);
        } while (v < target);
    }
    __syncthreads();
}

// ---------------------------------------------------------------------------
// Persistent recurrent kernel: one launch runs all T timesteps of a layer.
// 8 blocks x 128 thr (4 waves). Wave owns 16 h-columns across all 4 gates.
// Whh slice for the block (4 waves * 4 gates * 16 rows * 512 = 256KB bf16)
// is staged in LDS once; h_{t-1} (32KB bf16) is staged in LDS every step.
// Total LDS = 288KB <= 320KB/WGP -> one block per WGP.
// ---------------------------------------------------------------------------
__global__ __launch_bounds__(128) void lstm_layer_kernel(
    const unsigned short* __restrict__ Whh,     // [2048][512] bf16
    const float* __restrict__ Gx,               // [T][32][2048] fp32 (incl. bias)
    unsigned short* __restrict__ hbfA,          // [32][512] bf16 (t even: in)
    unsigned short* __restrict__ hbfB,          // [32][512] bf16 (t even: out)
    float* __restrict__ hstate,                 // [32][512] fp32
    float* __restrict__ cstate,                 // [32][512] fp32
    float* __restrict__ out_f32,                // null or [B][T][H]
    unsigned short* __restrict__ out_bf,        // null or [T][B][H]
    unsigned int* __restrict__ sync_counter) {
    __shared__ __align__(16) unsigned short ldsW[256 * 512];  // 256 KB
    __shared__ __align__(16) unsigned short ldsH[32 * 512];   //  32 KB

    const int tid  = threadIdx.x;
    const int lane = tid & 31;
    const int g    = lane >> 4;
    const int ln   = lane & 15;
    const int wib  = tid >> 5;                              // wave in block 0..3
    const int ncol = (blockIdx.x * 4 + wib) * 16;           // 0..511 step 16

    // ---- stage this block's Whh slice into LDS (once per layer) ----
    // LDS row r = (w*4 + q)*16 + c  <->  global row j = q*512 + (blk*4+w)*16 + c
    for (int c0 = tid; c0 < 16384; c0 += 128) {             // 16B chunks
        const int r  = c0 >> 6;                             // 0..255
        const int k8 = (c0 & 63) * 8;                       // ushort offset in row
        const int c  = r & 15;
        const int q  = (r >> 4) & 3;
        const int w  = r >> 6;
        const int j  = q * HID + (blockIdx.x * 4 + w) * 16 + c;
        *(u32x4*)(&ldsW[r * K_DIM + k8]) =
            *(const u32x4*)(Whh + (size_t)j * K_DIM + k8);
    }
    __syncthreads();

    for (int t = 0; t < T_STEPS; ++t) {
        const unsigned short* hin = (t & 1) ? hbfB : hbfA;
        unsigned short*      hout = (t & 1) ? hbfA : hbfB;
        const float* Gx_t = Gx + (size_t)t * BATCH * GATES;

        // warm L2->WGP path for this wave's Gx[t] lines (read in epilogue)
#pragma unroll
        for (int p = 0; p < 4; ++p) {
            const int idx = lane + 32 * p;                  // 0..127
            const int b   = idx & 31;
            const int q   = idx >> 5;
            __builtin_prefetch(Gx_t + (size_t)b * GATES + q * HID + ncol, 0, 0);
        }

        // ---- stage h_{t-1} into LDS ----
        for (int c0 = tid; c0 < 2048; c0 += 128) {          // 16B chunks
            const int r  = c0 >> 6;                         // batch row 0..31
            const int k8 = (c0 & 63) * 8;
            *(u32x4*)(&ldsH[r * HID + k8]) =
                *(const u32x4*)(hin + (size_t)r * HID + k8);
        }
        __syncthreads();

        // ---- gates = h_{t-1} * Whh^T (WMMA, all operands in LDS) ----
        v8f acc[4][2];
#pragma unroll
        for (int q = 0; q < 4; ++q)
#pragma unroll
            for (int mt = 0; mt < 2; ++mt)
#pragma unroll
                for (int r = 0; r < 8; ++r) acc[q][mt][r] = 0.0f;

        const unsigned short* a0row = &ldsH[ln * HID];         // rows 0..15
        const unsigned short* a1row = &ldsH[(16 + ln) * HID];  // rows 16..31
#pragma unroll 4
        for (int kk = 0; kk < HID / 32; ++kk) {
            const int kbase = kk * 32;
            v16bf a0 = load_a_frag(a0row, kbase, g);
            v16bf a1 = load_a_frag(a1row, kbase, g);
#pragma unroll
            for (int q = 0; q < 4; ++q) {
                const unsigned short* wrow = &ldsW[((wib * 4 + q) * 16 + ln) * K_DIM];
                v16bf bf = load_b_frag(wrow, kbase, g);
                acc[q][0] = wmma_bf16(a0, bf, acc[q][0]);
                acc[q][1] = wmma_bf16(a1, bf, acc[q][1]);
            }
        }

        // ---- LSTM cell update (elementwise, local to this wave) ----
        const int n = ncol + ln;
#pragma unroll
        for (int mt = 0; mt < 2; ++mt) {
#pragma unroll
            for (int r = 0; r < 8; ++r) {
                const int b = mt * 16 + 8 * g + r;
                const float* gb = Gx_t + (size_t)b * GATES;
                float xi = acc[0][mt][r] + gb[0 * HID + n];
                float xf = acc[1][mt][r] + gb[1 * HID + n];
                float xg = acc[2][mt][r] + gb[2 * HID + n];
                float xo = acc[3][mt][r] + gb[3 * HID + n];
                float ig = 1.0f / (1.0f + __expf(-xi));
                float fg = 1.0f / (1.0f + __expf(-xf));
                float gg = tanhf(xg);
                float og = 1.0f / (1.0f + __expf(-xo));
                float c_old = cstate[(size_t)b * HID + n];
                float c_new = fg * c_old + ig * gg;
                float h_new = og * tanhf(c_new);
                cstate[(size_t)b * HID + n] = c_new;
                hstate[(size_t)b * HID + n] = h_new;
                hout[(size_t)b * HID + n]   = f2bf(h_new);
                if (out_f32) out_f32[((size_t)b * T_STEPS + t) * HID + n] = h_new;
                if (out_bf)  out_bf[((size_t)t * BATCH + b) * HID + n] = f2bf(h_new);
            }
        }

        grid_sync(sync_counter, (unsigned int)(NBLK * (t + 1)));
    }
}

// ----------------------------- helper kernels ------------------------------
__global__ void convert_bf_kernel(unsigned short* __restrict__ dst,
                                  const float* __restrict__ src, int n) {
    int i = blockIdx.x * blockDim.x + threadIdx.x;
    if (i < n) dst[i] = f2bf(src[i]);
}

__global__ void bias_kernel(float* __restrict__ dst, const float* __restrict__ bih,
                            const float* __restrict__ bhh, int n) {
    int i = blockIdx.x * blockDim.x + threadIdx.x;
    if (i < n) dst[i] = bih[i] + bhh[i];
}

// x: [B][T][I] fp32 -> dst: [T][B][I] bf16  (rows r = t*32+b)
__global__ void convert_x_kernel(unsigned short* __restrict__ dst,
                                 const float* __restrict__ src) {
    int idx = blockIdx.x * blockDim.x + threadIdx.x;   // < T*B*I
    int i = idx & (HID - 1);
    int r = idx >> 9;
    int b = r & (BATCH - 1);
    int t = r >> 5;
    dst[idx] = f2bf(src[((size_t)b * T_STEPS + t) * HID + i]);
}

__global__ void init_state_kernel(float* __restrict__ hstate, float* __restrict__ cstate,
                                  unsigned short* __restrict__ hbf,
                                  const float* __restrict__ h0,
                                  const float* __restrict__ c0,
                                  unsigned int* __restrict__ sync_counter, int n) {
    int i = blockIdx.x * blockDim.x + threadIdx.x;
    if (i == 0) *sync_counter = 0u;
    if (i < n) {
        float h = h0[i];
        hstate[i] = h;
        hbf[i]    = f2bf(h);
        cstate[i] = c0[i];
    }
}

__global__ void finalize_kernel(float* __restrict__ out_hc,
                                const float* __restrict__ hstate,
                                const float* __restrict__ cstate, int n) {
    int i = blockIdx.x * blockDim.x + threadIdx.x;
    if (i < n) {
        out_hc[i]     = hstate[i];
        out_hc[n + i] = cstate[i];
    }
}

// ---------------------------------------------------------------------------
extern "C" void kernel_launch(void* const* d_in, const int* in_sizes, int n_in,
                              void* d_out, int out_size, void* d_ws, size_t ws_size,
                              hipStream_t stream) {
    const float* x   = (const float*)d_in[0];   // [B,T,I]
    const float* h0  = (const float*)d_in[1];   // [L,B,H]
    const float* c0  = (const float*)d_in[2];   // [L,B,H]
    const float* Wih = (const float*)d_in[3];   // [L,4H,I]
    const float* Whh = (const float*)d_in[4];   // [L,4H,H]
    const float* bih = (const float*)d_in[5];   // [L,4H]
    const float* bhh = (const float*)d_in[6];   // [L,4H]
    float* out = (float*)d_out;

    char* ws = (char*)d_ws;
    size_t off = 0;
    auto alloc = [&](size_t bytes) {
        void* p = ws + off;
        off = (off + bytes + 255) & ~(size_t)255;
        return p;
    };
    unsigned short* WihBf = (unsigned short*)alloc((size_t)2 * GATES * K_DIM * 2);
    unsigned short* WhhBf = (unsigned short*)alloc((size_t)2 * GATES * K_DIM * 2);
    float*          Bias  = (float*)alloc((size_t)2 * GATES * 4);
    unsigned short* Xbf0  = (unsigned short*)alloc((size_t)T_STEPS * BATCH * HID * 2);
    unsigned short* X1bf  = (unsigned short*)alloc((size_t)T_STEPS * BATCH * HID * 2);
    float*          Gx    = (float*)alloc((size_t)T_STEPS * BATCH * GATES * 4);
    float*          Hs    = (float*)alloc((size_t)2 * BATCH * HID * 4);
    float*          Cs    = (float*)alloc((size_t)2 * BATCH * HID * 4);
    unsigned short* hbfA  = (unsigned short*)alloc((size_t)BATCH * HID * 2);
    unsigned short* hbfB  = (unsigned short*)alloc((size_t)BATCH * HID * 2);
    unsigned int*   Sync  = (unsigned int*)alloc(2 * sizeof(unsigned int) + 256);

    const int nw = 2 * GATES * K_DIM;
    convert_bf_kernel<<<(nw + 255) / 256, 256, 0, stream>>>(WihBf, Wih, nw);
    convert_bf_kernel<<<(nw + 255) / 256, 256, 0, stream>>>(WhhBf, Whh, nw);
    bias_kernel<<<(2 * GATES + 255) / 256, 256, 0, stream>>>(Bias, bih, bhh, 2 * GATES);
    const int nx = T_STEPS * BATCH * HID;
    convert_x_kernel<<<nx / 256, 256, 0, stream>>>(Xbf0, x);

    for (int l = 0; l < 2; ++l) {
        init_state_kernel<<<(BATCH * HID + 255) / 256, 256, 0, stream>>>(
            Hs + (size_t)l * BATCH * HID, Cs + (size_t)l * BATCH * HID, hbfA,
            h0 + (size_t)l * BATCH * HID, c0 + (size_t)l * BATCH * HID,
            Sync + l, BATCH * HID);

        const unsigned short* Ain = (l == 0) ? Xbf0 : X1bf;
        dim3 ggrid(GATES / 64, (T_STEPS * BATCH) / 128);
        gemm_gates_kernel<<<ggrid, 256, 0, stream>>>(
            Ain, WihBf + (size_t)l * GATES * K_DIM, Bias + l * GATES, Gx);

        lstm_layer_kernel<<<NBLK, 128, 0, stream>>>(
            WhhBf + (size_t)l * GATES * K_DIM, Gx, hbfA, hbfB,
            Hs + (size_t)l * BATCH * HID, Cs + (size_t)l * BATCH * HID,
            (l == 1) ? out : nullptr,
            (l == 0) ? X1bf : nullptr,
            Sync + l);
    }
    finalize_kernel<<<(2 * BATCH * HID + 255) / 256, 256, 0, stream>>>(
        out + (size_t)BATCH * T_STEPS * HID, Hs, Cs, 2 * BATCH * HID);
}
